// DMSTGCN_12996571037726
// MI455X (gfx1250) — matrix operand order repaired and checked by reading, not compiled, not run on hardware
//
#include <hip/hip_runtime.h>
#include <math.h>

typedef __bf16 bf16_t;
typedef __attribute__((ext_vector_type(16))) __bf16 v16bf;
typedef __attribute__((ext_vector_type(8)))  __bf16 v8bf;
typedef __attribute__((ext_vector_type(8)))  float  v8f;

#define B_    16
#define N_    1024
#define T_    12
#define RF_   13
#define RC_   16
#define SC_   8
#define DIMS_ 32
#define NL_   8
#define CL_   (RC_*RF_)        // 208 = channel*time rows for WMMA "N" dim
#define SKC_  (SC_*RF_*NL_)    // 832 skip channels

static __device__ __forceinline__ bf16_t f2bf(float f) {
  unsigned u = __builtin_bit_cast(unsigned, f);
  unsigned r = u + 0x7FFFu + ((u >> 16) & 1u);   // round-to-nearest-even
  unsigned short h = (unsigned short)(r >> 16);
  return __builtin_bit_cast(bf16_t, h);
}
static __device__ __forceinline__ float bf2f(bf16_t b) {
  unsigned short h = __builtin_bit_cast(unsigned short, b);
  return __builtin_bit_cast(float, ((unsigned)h) << 16);
}

// ---------------------------------------------------------------------------
// Setup: start 1x1 convs with left zero-pad of 1 timestep.
// x layout: [b][l][c][n]  (n contiguous -> coalesced everywhere)
// ---------------------------------------------------------------------------
__global__ void k_start(const float* __restrict__ inp,
                        const float* __restrict__ sw, const float* __restrict__ sb,
                        const float* __restrict__ saw, const float* __restrict__ sab,
                        float* __restrict__ xcur, float* __restrict__ xa) {
  int idx = blockIdx.x * 256 + threadIdx.x;      // ((b*13+l)*16+c)*1024+n
  int n = idx & (N_ - 1);
  int c = (idx >> 10) & 15;
  int l = (idx >> 14) % RF_;
  int b = idx / (RF_ * 16 * N_);
  float v0 = 0.f, v1 = 0.f;
  if (l > 0) {
    size_t base = ((size_t)(b * 2 + 0) * N_ + n) * T_ + (l - 1);
    v0 = inp[base];
    v1 = inp[base + (size_t)N_ * T_];
  }
  xcur[idx] = sw[c] * v0 + sb[c];
  xa[idx]   = saw[c] * v1 + sab[c];
}

// adp[b][j][k] = sum_i p1[ind[b]][i] * pk[i][j][k]
__global__ void k_adp(const float* __restrict__ p1, const int* __restrict__ ind,
                      const float* __restrict__ pk, float* __restrict__ adp) {
  int idx = blockIdx.x * 256 + threadIdx.x;      // (b*32+j)*32+k
  int jk = idx & 1023;
  int b  = idx >> 10;
  const float* te = p1 + (size_t)ind[b] * DIMS_;
  float acc = 0.f;
  #pragma unroll
  for (int i = 0; i < DIMS_; ++i) acc += te[i] * pk[(size_t)i * 1024 + jk];
  adp[idx] = acc;
}

// src[b][n][k] = sum_j p2[n][j] * adp[b][j][k]
__global__ void k_src(const float* __restrict__ p2, const float* __restrict__ adp,
                      float* __restrict__ src) {
  int idx = blockIdx.x * 256 + threadIdx.x;      // ((b*1024)+n)*32+k
  int k = idx & 31;
  int n = (idx >> 5) & (N_ - 1);
  int b = idx >> 15;
  const float* pr = p2 + (size_t)n * DIMS_;
  const float* ad = adp + (size_t)b * 1024 + k;
  float acc = 0.f;
  #pragma unroll
  for (int j = 0; j < DIMS_; ++j) acc += pr[j] * ad[j * 32];
  src[idx] = acc;
}

// temp[b][n][c] = sum_k p3[c][k] * src[b][n][k]   (only first 32 rows of p3 used)
__global__ void k_temp(const float* __restrict__ p3, const float* __restrict__ src,
                       float* __restrict__ temp) {
  int idx = blockIdx.x * 256 + threadIdx.x;      // ((b*1024)+n)*32+c
  int c  = idx & 31;
  int bn = idx >> 5;
  const float* sp  = src + (size_t)bn * DIMS_;
  const float* p3r = p3 + (size_t)c * DIMS_;
  float acc = 0.f;
  #pragma unroll
  for (int k = 0; k < DIMS_; ++k) acc += p3r[k] * sp[k];
  temp[idx] = acc;
}

// A[b][w][v] = relu(tanh(src[w].temp[v] - src[v].temp[w])), bf16 row-major [w][v]
__global__ void k_adj(const float* __restrict__ src, const float* __restrict__ temp,
                      bf16_t* __restrict__ A) {
  __shared__ float s_sw[16][32], s_tw[16][32], s_sv[16][32], s_tv[16][32];
  int b  = blockIdx.z;
  int w0 = blockIdx.y * 16;
  int v0 = blockIdx.x * 16;
  int tid = threadIdx.x;
  for (int i = tid; i < 512; i += 256) {
    int r = i >> 5, k = i & 31;
    s_sw[r][k] = src [((size_t)b * N_ + w0 + r) * DIMS_ + k];
    s_tw[r][k] = temp[((size_t)b * N_ + w0 + r) * DIMS_ + k];
    s_sv[r][k] = src [((size_t)b * N_ + v0 + r) * DIMS_ + k];
    s_tv[r][k] = temp[((size_t)b * N_ + v0 + r) * DIMS_ + k];
  }
  __syncthreads();
  int i = tid >> 4;   // local w
  int j = tid & 15;   // local v (fastest -> coalesced store)
  float d1 = 0.f, d2 = 0.f;
  #pragma unroll
  for (int k = 0; k < DIMS_; ++k) {
    d1 += s_sw[i][k] * s_tv[j][k];
    d2 += s_sv[j][k] * s_tw[i][k];
  }
  float a = tanhf(d1 - d2);
  A[((size_t)b * N_ + w0 + i) * N_ + v0 + j] = f2bf(fmaxf(a, 0.f));
}

// ---------------------------------------------------------------------------
// Per-layer fused fc1/relu/fc2/sigmoid + skip conv.
// Writes Xt[b][c*13+l][n] bf16 (B-operand layout) and relu'd skip channels.
// ---------------------------------------------------------------------------
__global__ void k_fc(const float* __restrict__ xcur,
                     const float* __restrict__ fc1, const float* __restrict__ fc2,
                     const float* __restrict__ skw, const float* __restrict__ skb,
                     bf16_t* __restrict__ Xt, float* __restrict__ skipbuf,
                     int skip_off) {
  int idx = blockIdx.x * 256 + threadIdx.x;      // (b*13+l)*1024+n
  int n = idx & (N_ - 1);
  int l = (idx >> 10) % RF_;
  int b = idx / (RF_ * N_);
  size_t base = ((size_t)(b * RF_ + l) * 16) * N_ + n;
  float xc[16], t[16], xs[16];
  #pragma unroll
  for (int c = 0; c < 16; ++c) xc[c] = xcur[base + (size_t)c * N_];
  #pragma unroll
  for (int o = 0; o < 16; ++o) {
    float a = 0.f;
    #pragma unroll
    for (int c = 0; c < 16; ++c) a += fc1[o * 16 + c] * xc[c];
    t[o] = fmaxf(a, 0.f);
  }
  #pragma unroll
  for (int o = 0; o < 16; ++o) {
    float a = 0.f;
    #pragma unroll
    for (int c = 0; c < 16; ++c) a += fc2[o * 16 + c] * t[c];
    float z = 2.f * a + xc[o];
    xs[o] = 1.f / (1.f + __expf(-z));
  }
  size_t tb = (size_t)b * CL_ * N_ + (size_t)l * N_ + n;
  #pragma unroll
  for (int c = 0; c < 16; ++c) Xt[tb + (size_t)c * RF_ * N_] = f2bf(xs[c]);
  #pragma unroll
  for (int o = 0; o < SC_; ++o) {
    float a = skb[o];
    #pragma unroll
    for (int c = 0; c < 16; ++c) a += skw[o * 16 + c] * xs[c];
    skipbuf[((size_t)b * SKC_ + skip_off + o * RF_ + l) * N_ + n] = fmaxf(a, 0.f);
  }
}

// ---------------------------------------------------------------------------
// WMMA GEMM: Out[n][m] = sum_k A[m][k] * Bt[n][k]    (per batch b = blockIdx.z)
// A: [1024][1024] bf16 row-major (M x K).  Bt: [208][1024] bf16 (N x K).
// Each wave: 64(M) x 16(N) strip; K=1024, ping-pong double-buffered so the
// loads for k-step i+1 are in flight while the 4 WMMAs of k-step i execute
// (avoids the s_wait_loadcnt 0 full-drain before every WMMA).
// A-operand lane layout (ISA 7.12.2): lanes 0-15 hold K{0..7,16..23} of row m,
// lanes 16-31 hold K{8..15,24..31}. B-operand: same K split as A, column n.
// ---------------------------------------------------------------------------
static __device__ __forceinline__ v16bf load_a16(const bf16_t* p) {
  v8bf lo = *(const v8bf*)(p);        // K .. K+7
  v8bf hi = *(const v8bf*)(p + 16);   // K+16 .. K+23
  v16bf r;
  #pragma unroll
  for (int i = 0; i < 8; ++i) { r[i] = lo[i]; r[i + 8] = hi[i]; }
  return r;
}

__global__ void __launch_bounds__(128) k_gemm(const bf16_t* __restrict__ Abuf,
                                              const bf16_t* __restrict__ Bt,
                                              bf16_t* __restrict__ Out) {
  const int K = N_;
  int b = blockIdx.z;
  const bf16_t* A  = Abuf + (size_t)b * N_ * N_;
  const bf16_t* Bp = Bt   + (size_t)b * CL_ * N_;
  bf16_t*       Op = Out  + (size_t)b * CL_ * N_;
  int wave = threadIdx.x >> 5;
  int lane = threadIdx.x & 31;
  int hi   = lane >> 4;          // 0 or 1 (lane half)
  int lo   = lane & 15;
  int m_base = (blockIdx.x * 4 + wave) * 64;
  int ncol   = blockIdx.y * 16 + lo;
  const bf16_t* pb = Bp + (size_t)ncol * K + hi * 8;
  const bf16_t* pa = A  + (size_t)(m_base + lo) * K + hi * 8;
  v8f c0 = {}, c1 = {}, c2 = {}, c3 = {};

  // prologue: buffer set 0 <- k=0
  v16bf b0  = load_a16(pb);
  v16bf a00 = load_a16(pa);
  v16bf a01 = load_a16(pa + (size_t)16 * K);
  v16bf a02 = load_a16(pa + (size_t)32 * K);
  v16bf a03 = load_a16(pa + (size_t)48 * K);

  for (int k0 = 0; k0 < K; k0 += 64) {
    // prefetch buffer set 1 <- k0+32 while computing on set 0
    int k1 = k0 + 32;
    v16bf b1  = load_a16(pb + k1);
    v16bf a10 = load_a16(pa + k1);
    v16bf a11 = load_a16(pa + (size_t)16 * K + k1);
    v16bf a12 = load_a16(pa + (size_t)32 * K + k1);
    v16bf a13 = load_a16(pa + (size_t)48 * K + k1);
    c0 = __builtin_amdgcn_wmma_f32_16x16x32_bf16(false, a00, false, b0, (short)0, c0, false, false);
    c1 = __builtin_amdgcn_wmma_f32_16x16x32_bf16(false, a01, false, b0, (short)0, c1, false, false);
    c2 = __builtin_amdgcn_wmma_f32_16x16x32_bf16(false, a02, false, b0, (short)0, c2, false, false);
    c3 = __builtin_amdgcn_wmma_f32_16x16x32_bf16(false, a03, false, b0, (short)0, c3, false, false);
    // prefetch buffer set 0 <- k0+64 (wraps to 0 on last iter; dead but in-bounds)
    int k2 = (k0 + 64) & (K - 1);
    b0  = load_a16(pb + k2);
    a00 = load_a16(pa + k2);
    a01 = load_a16(pa + (size_t)16 * K + k2);
    a02 = load_a16(pa + (size_t)32 * K + k2);
    a03 = load_a16(pa + (size_t)48 * K + k2);
    c0 = __builtin_amdgcn_wmma_f32_16x16x32_bf16(false, a10, false, b1, (short)0, c0, false, false);
    c1 = __builtin_amdgcn_wmma_f32_16x16x32_bf16(false, a11, false, b1, (short)0, c1, false, false);
    c2 = __builtin_amdgcn_wmma_f32_16x16x32_bf16(false, a12, false, b1, (short)0, c2, false, false);
    c3 = __builtin_amdgcn_wmma_f32_16x16x32_bf16(false, a13, false, b1, (short)0, c3, false, false);
  }

  // D layout: element r of v8f -> row m_tile + r + 8*hi, col = lo.
  int mh = hi * 8;
  v8f cc[4] = {c0, c1, c2, c3};
  #pragma unroll
  for (int t = 0; t < 4; ++t) {
    v8bf d;
    #pragma unroll
    for (int r = 0; r < 8; ++r) d[r] = f2bf(cc[t][r]);
    *(v8bf*)(Op + (size_t)ncol * N_ + m_base + t * 16 + mh) = d;
  }
}

// ---------------------------------------------------------------------------
// Per-layer fused gconv (16x48) + x_a + residual + both batch norms.
// ---------------------------------------------------------------------------
__global__ void k_gconv(const bf16_t* __restrict__ Xt, const bf16_t* __restrict__ H1t,
                        const bf16_t* __restrict__ H2t,
                        float* __restrict__ xcur, float* __restrict__ xa,
                        const float* __restrict__ gw, const float* __restrict__ gb,
                        const float* __restrict__ bng, const float* __restrict__ bnb,
                        const float* __restrict__ bnag, const float* __restrict__ bnab) {
  int idx = blockIdx.x * 256 + threadIdx.x;      // (b*13+l)*1024+n
  int n = idx & (N_ - 1);
  int l = (idx >> 10) % RF_;
  int b = idx / (RF_ * N_);
  float acc[16];
  #pragma unroll
  for (int o = 0; o < 16; ++o) acc[o] = gb[o];
  size_t tb = (size_t)b * CL_ * N_ + (size_t)l * N_ + n;
  #pragma unroll
  for (int c = 0; c < 16; ++c) {
    size_t off = tb + (size_t)c * RF_ * N_;
    float xv = bf2f(Xt[off]);
    float h1 = bf2f(H1t[off]);
    float h2 = bf2f(H2t[off]);
    #pragma unroll
    for (int o = 0; o < 16; ++o)
      acc[o] += gw[o * 48 + c] * xv + gw[o * 48 + 16 + c] * h1 + gw[o * 48 + 32 + c] * h2;
  }
  const float BNS = 0.9999950000374997f;   // 1/sqrt(1+1e-5)
  size_t base = ((size_t)(b * RF_ + l) * 16) * N_ + n;
  #pragma unroll
  for (int o = 0; o < 16; ++o) {
    float res = xcur[base + (size_t)o * N_];
    float xav = xa  [base + (size_t)o * N_];
    xcur[base + (size_t)o * N_] = (acc[o] + xav + res) * (bng[o] * BNS) + bnb[o];
    xa  [base + (size_t)o * N_] = (2.f * xav) * (bnag[o] * BNS) + bnab[o];
  }
}

// ---------------------------------------------------------------------------
// End MLP stage 1: out1[b][oc][n] = relu(sum_k end1_w[oc][k]*skip[b][k][n] + b)
// LDS-tiled over 32 k x 64 n.
// ---------------------------------------------------------------------------
__global__ void k_end1(const float* __restrict__ skip, const float* __restrict__ w,
                       const float* __restrict__ bias, float* __restrict__ out1) {
  __shared__ float tile[32][64];
  int b  = blockIdx.y;
  int n0 = blockIdx.x * 64;
  int tid = threadIdx.x;
  int oc = tid & 63;
  int ng = tid >> 6;           // 0..3, each owns 16 n
  float acc[16];
  #pragma unroll
  for (int t = 0; t < 16; ++t) acc[t] = 0.f;
  for (int kc = 0; kc < SKC_; kc += 32) {
    for (int i = tid; i < 32 * 64; i += 256) {
      int k = i >> 6, n = i & 63;
      tile[k][n] = skip[((size_t)b * SKC_ + kc + k) * N_ + n0 + n];
    }
    __syncthreads();
    #pragma unroll 8
    for (int k = 0; k < 32; ++k) {
      float wv = w[(size_t)oc * SKC_ + kc + k];
      #pragma unroll
      for (int t = 0; t < 16; ++t) acc[t] += wv * tile[k][ng * 16 + t];
    }
    __syncthreads();
  }
  #pragma unroll
  for (int t = 0; t < 16; ++t)
    out1[((size_t)b * 64 + oc) * N_ + n0 + ng * 16 + t] = fmaxf(acc[t] + bias[oc], 0.f);
}

// End MLP stage 2: out[b][oc][n] = sum_k end2_w[oc][k]*out1[b][k][n] + b
__global__ void k_end2(const float* __restrict__ out1, const float* __restrict__ w,
                       const float* __restrict__ bias, float* __restrict__ out) {
  int idx = blockIdx.x * 256 + threadIdx.x;      // ((b*12)+oc)*1024+n
  int n  = idx & (N_ - 1);
  int oc = (idx >> 10) % 12;
  int b  = idx / (12 * N_);
  float acc = bias[oc];
  #pragma unroll
  for (int k = 0; k < 64; ++k)
    acc += w[oc * 64 + k] * out1[((size_t)b * 64 + k) * N_ + n];
  out[idx] = acc;
}

// ---------------------------------------------------------------------------
extern "C" void kernel_launch(void* const* d_in, const int* in_sizes, int n_in,
                              void* d_out, int out_size, void* d_ws, size_t ws_size,
                              hipStream_t stream) {
  (void)in_sizes; (void)n_in; (void)out_size; (void)ws_size;
  const float* inp     = (const float*)d_in[0];
  const int*   ind     = (const int*)  d_in[1];
  const float* start_w = (const float*)d_in[2];
  const float* start_b = (const float*)d_in[3];
  const float* starta_w= (const float*)d_in[4];
  const float* starta_b= (const float*)d_in[5];
  const float* p1      = (const float*)d_in[6];
  const float* p2      = (const float*)d_in[7];
  const float* p3      = (const float*)d_in[8];
  const float* pk      = (const float*)d_in[9];
  const float* fc1_w   = (const float*)d_in[10];
  const float* fc2_w   = (const float*)d_in[11];
  const float* skip_w  = (const float*)d_in[12];
  const float* skip_b  = (const float*)d_in[13];
  const float* gconv_w = (const float*)d_in[14];
  const float* gconv_b = (const float*)d_in[15];
  const float* bn_g    = (const float*)d_in[16];
  const float* bn_b    = (const float*)d_in[17];
  const float* bna_g   = (const float*)d_in[18];
  const float* bna_b   = (const float*)d_in[19];
  const float* end1_w  = (const float*)d_in[20];
  const float* end1_b  = (const float*)d_in[21];
  const float* end2_w  = (const float*)d_in[22];
  const float* end2_b  = (const float*)d_in[23];

  char* ws = (char*)d_ws;
  size_t off = 0;
  auto suballoc = [&](size_t bytes) -> void* {
    void* p = ws + off;
    off = (off + bytes + 255) & ~(size_t)255;
    return p;
  };
  float*  xcur   = (float*) suballoc((size_t)B_ * RF_ * 16 * N_ * 4);   // 13.6 MB
  float*  xa     = (float*) suballoc((size_t)B_ * RF_ * 16 * N_ * 4);
  bf16_t* Xt     = (bf16_t*)suballoc((size_t)B_ * CL_ * N_ * 2);        // 6.8 MB
  bf16_t* H1t    = (bf16_t*)suballoc((size_t)B_ * CL_ * N_ * 2);
  bf16_t* H2t    = (bf16_t*)suballoc((size_t)B_ * CL_ * N_ * 2);
  bf16_t* Abf    = (bf16_t*)suballoc((size_t)B_ * N_ * N_ * 2);         // 33.5 MB
  float*  srcb   = (float*) suballoc((size_t)B_ * N_ * DIMS_ * 4);
  float*  tempb  = (float*) suballoc((size_t)B_ * N_ * DIMS_ * 4);
  float*  adpb   = (float*) suballoc((size_t)B_ * DIMS_ * DIMS_ * 4);
  float*  skipbf = (float*) suballoc((size_t)B_ * SKC_ * N_ * 4);       // 54.5 MB
  float*  out1   = (float*) suballoc((size_t)B_ * 64 * N_ * 4);

  // --- setup ---
  k_start<<<(B_ * RF_ * 16 * N_) / 256, 256, 0, stream>>>(inp, start_w, start_b,
                                                          starta_w, starta_b, xcur, xa);
  k_adp <<<(B_ * DIMS_ * DIMS_) / 256, 256, 0, stream>>>(p1, ind, pk, adpb);
  k_src <<<(B_ * N_ * DIMS_) / 256, 256, 0, stream>>>(p2, adpb, srcb);
  k_temp<<<(B_ * N_ * DIMS_) / 256, 256, 0, stream>>>(p3, srcb, tempb);
  k_adj <<<dim3(N_ / 16, N_ / 16, B_), 256, 0, stream>>>(srcb, tempb, Abf);

  // --- layers ---
  for (int i = 0; i < NL_; ++i) {
    k_fc<<<(B_ * RF_ * N_) / 256, 256, 0, stream>>>(
        xcur, fc1_w, fc2_w, skip_w + (size_t)i * SC_ * RC_, skip_b + (size_t)i * SC_,
        Xt, skipbf, (NL_ - 1 - i) * SC_ * RF_);
    k_gemm<<<dim3(4, CL_ / 16, B_), 128, 0, stream>>>(Abf, Xt, H1t);
    k_gemm<<<dim3(4, CL_ / 16, B_), 128, 0, stream>>>(Abf, H1t, H2t);
    k_gconv<<<(B_ * RF_ * N_) / 256, 256, 0, stream>>>(
        Xt, H1t, H2t, xcur, xa,
        gconv_w + (size_t)i * RC_ * 3 * RC_, gconv_b + (size_t)i * RC_,
        bn_g + (size_t)i * RC_, bn_b + (size_t)i * RC_,
        bna_g + (size_t)i * RC_, bna_b + (size_t)i * RC_);
  }

  // --- end MLP ---
  k_end1<<<dim3(N_ / 64, B_), 256, 0, stream>>>(skipbf, end1_w, end1_b, out1);
  k_end2<<<(B_ * 12 * N_) / 256, 256, 0, stream>>>(out1, end2_w, end2_b, (float*)d_out);
}
